// LinkPredictionGNN_53558242181177
// MI455X (gfx1250) — compile-verified, matrix operand order checked by reading, not run on hardware
//
#include <hip/hip_runtime.h>
#include <hip/hip_bf16.h>

// ---------------- types ----------------
typedef __attribute__((ext_vector_type(16))) __bf16       v16bf;
typedef __attribute__((ext_vector_type(8)))  float        v8f;
typedef __attribute__((ext_vector_type(8)))  unsigned int v8u;

#define IN_CH  128
#define HID_CH 128
#define OUT_CH 64

static __device__ __forceinline__ unsigned short f2bf(float f) {
    unsigned int u = __builtin_bit_cast(unsigned int, f);
    unsigned int r = u + 0x7FFFu + ((u >> 16) & 1u);   // round-to-nearest-even
    return (unsigned short)(r >> 16);
}

// ---------------- degree / norm ----------------
__global__ void k_deg_init(float* __restrict__ deg, int n) {
    int i = blockIdx.x * blockDim.x + threadIdx.x;
    if (i < n) deg[i] = 1.0f;                // self-loop
}

__global__ void k_deg_count(const long long* __restrict__ dst, float* __restrict__ deg, int e) {
    int i = blockIdx.x * blockDim.x + threadIdx.x;
    if (i < e) atomicAdd(&deg[(int)dst[i]], 1.0f);
}

__global__ void k_dinv(float* __restrict__ deg, int n) {
    int i = blockIdx.x * blockDim.x + threadIdx.x;
    if (i < n) {
        float d = deg[i];
        deg[i] = (d > 0.f) ? rsqrtf(d) : 0.f;
    }
}

// ---------------- conversions ----------------
__global__ void k_cvt_bf16(const float* __restrict__ in, unsigned short* __restrict__ out, size_t n) {
    size_t i = (size_t)blockIdx.x * blockDim.x + threadIdx.x;
    if (i < n) out[i] = f2bf(in[i]);
}

// W [K, Ncols] f32  ->  Wt [Ncols, K] bf16  (transpose so K is contiguous)
__global__ void k_cvt_wT(const float* __restrict__ W, unsigned short* __restrict__ Wt, int K, int Ncols) {
    int idx = blockIdx.x * blockDim.x + threadIdx.x;
    if (idx < K * Ncols) {
        int k = idx / Ncols, n = idx % Ncols;
        Wt[(size_t)n * K + k] = f2bf(W[idx]);
    }
}

// ---------------- WMMA GEMM:  C[M,Ncols] = A[M,K](bf16) x Bt[Ncols,K](bf16)^T ----------------
// Register-blocked: one wave computes a 64x32 output tile (4 M-tiles x 2 N-tiles).
// Per K-step: 4 A-fragment loads + 2 B-fragment loads -> 8 WMMAs (B reused across 4 row tiles).
// blockDim.x = 32 * (Ncols/32); gridDim.x = ceil(M/64)
__global__ void k_gemm_bf16(const unsigned short* __restrict__ A,
                            const unsigned short* __restrict__ Bt,
                            float* __restrict__ C, int M, int K, int Ncols) {
    int lane  = threadIdx.x & 31;
    int wv    = threadIdx.x >> 5;
    int row0  = blockIdx.x * 64;
    int col0  = wv * 32;
    int half  = lane >> 4;          // which half of the wave
    int idx15 = lane & 15;          // M index (for A) / N index (for B)

    const unsigned short* Arow[4];
#pragma unroll
    for (int mt = 0; mt < 4; ++mt) {
        int r = row0 + 16 * mt + idx15;
        if (r >= M) r = M - 1;      // clamp tail rows (stores are guarded)
        Arow[mt] = A + (size_t)r * K;
    }
    const unsigned short* Brow0 = Bt + (size_t)(col0 + idx15) * K;
    const unsigned short* Brow1 = Bt + (size_t)(col0 + 16 + idx15) * K;

    v8f acc[4][2] = {};
    for (int kt = 0; kt < K; kt += 32) {
        // 16-bit fragment: vgpr vg holds a K-pair; lanes 0-15: K 0-7 & 16-23, lanes 16-31: +8
        v16bf a[4], b[2];
#pragma unroll
        for (int mt = 0; mt < 4; ++mt) {
            v8u au;
#pragma unroll
            for (int vg = 0; vg < 8; ++vg) {
                int kbase = kt + 2 * (vg & 3) + ((vg & 4) ? 16 : 0) + 8 * half;
                au[vg] = *(const unsigned int*)(Arow[mt] + kbase);
            }
            a[mt] = __builtin_bit_cast(v16bf, au);
        }
        {
            v8u bu0, bu1;
#pragma unroll
            for (int vg = 0; vg < 8; ++vg) {
                int kbase = kt + 2 * (vg & 3) + ((vg & 4) ? 16 : 0) + 8 * half;
                bu0[vg] = *(const unsigned int*)(Brow0 + kbase);
                bu1[vg] = *(const unsigned int*)(Brow1 + kbase);
            }
            b[0] = __builtin_bit_cast(v16bf, bu0);
            b[1] = __builtin_bit_cast(v16bf, bu1);
        }
#pragma unroll
        for (int mt = 0; mt < 4; ++mt)
#pragma unroll
            for (int nt = 0; nt < 2; ++nt)
                acc[mt][nt] = __builtin_amdgcn_wmma_f32_16x16x32_bf16(
                    false, a[mt], false, b[nt], (short)0, acc[mt][nt], false, false);
    }

    // C/D layout: vgpr r -> row (16*mt + r + 8*half), col = 16*nt + (lane&15)
#pragma unroll
    for (int mt = 0; mt < 4; ++mt) {
#pragma unroll
        for (int r = 0; r < 8; ++r) {
            int m = row0 + 16 * mt + r + 8 * half;
            if (m < M) {
#pragma unroll
                for (int nt = 0; nt < 2; ++nt)
                    C[(size_t)m * Ncols + col0 + 16 * nt + idx15] = acc[mt][nt][r];
            }
        }
    }
}

// ---------------- aggregation ----------------
// out[i][c] = t[i][c]*dinv[i]^2 + bias[c]     (self-loop term + bias, also zero-inits buffer)
__global__ void k_agg_init(const float* __restrict__ t, const float* __restrict__ dinv,
                           const float* __restrict__ bias, float* __restrict__ out,
                           int n, int C, int logC) {
    size_t idx = (size_t)blockIdx.x * blockDim.x + threadIdx.x;
    if (idx < (size_t)n * C) {
        int i = (int)(idx >> logC);
        int c = (int)(idx & (C - 1));
        float di = dinv[i];
        out[idx] = t[idx] * di * di + bias[c];
    }
}

// out[dst][c] += t[src][c] * dinv[src]*dinv[dst]   (one thread per (edge,channel))
__global__ void k_scatter(const float* __restrict__ t, const float* __restrict__ dinv,
                          const long long* __restrict__ ei, float* __restrict__ out,
                          int e, int C, int logC) {
    size_t idx = (size_t)blockIdx.x * blockDim.x + threadIdx.x;
    if (idx < (size_t)e * C) {
        int ed = (int)(idx >> logC);
        int c  = (int)(idx & (C - 1));
        int s  = (int)ei[ed];
        int d  = (int)ei[(size_t)e + ed];
        float nrm = dinv[s] * dinv[d];
        atomicAdd(&out[(size_t)d * C + c], t[(size_t)s * C + c] * nrm);
    }
}

// hb = bf16(relu(h))
__global__ void k_relu_cvt(const float* __restrict__ h, unsigned short* __restrict__ hb, size_t n) {
    size_t i = (size_t)blockIdx.x * blockDim.x + threadIdx.x;
    if (i < n) hb[i] = f2bf(fmaxf(h[i], 0.f));
}

// ---------------- decode: out[e] = dot(z[src[e]], z[dst[e]]) over 64 channels ----------------
__global__ void k_decode(const float* __restrict__ z, const long long* __restrict__ ei,
                         float* __restrict__ out, int e) {
    int i = blockIdx.x * blockDim.x + threadIdx.x;
    if (i >= e) return;
    long long s = ei[i];
    long long d = ei[(size_t)e + i];
    const float4* zs = (const float4*)(z + (size_t)s * OUT_CH);
    const float4* zd = (const float4*)(z + (size_t)d * OUT_CH);
    float acc = 0.f;
#pragma unroll
    for (int j = 0; j < OUT_CH / 4; ++j) {
        float4 a = zs[j], b = zd[j];
        acc += a.x * b.x + a.y * b.y + a.z * b.z + a.w * b.w;
    }
    out[i] = acc;
}

// ---------------- launcher ----------------
static inline size_t alignup(size_t v) { return (v + 255) & ~(size_t)255; }

extern "C" void kernel_launch(void* const* d_in, const int* in_sizes, int n_in,
                              void* d_out, int out_size, void* d_ws, size_t ws_size,
                              hipStream_t stream) {
    const float*     x   = (const float*)d_in[0];
    const long long* pei = (const long long*)d_in[1];
    const long long* nei = (const long long*)d_in[2];
    const float*     W1  = (const float*)d_in[3];
    const float*     b1  = (const float*)d_in[4];
    const float*     W2  = (const float*)d_in[5];
    const float*     b2  = (const float*)d_in[6];

    const int N = in_sizes[0] / IN_CH;      // 100000
    const int E = in_sizes[1] / 2;          // 1600000

    // workspace carve-out
    char* p = (char*)d_ws;
    float*          dinv = (float*)p;           p += alignup((size_t)N * 4);
    unsigned short* xb   = (unsigned short*)p;  p += alignup((size_t)N * IN_CH * 2);   // reused as hb
    unsigned short* W1t  = (unsigned short*)p;  p += alignup((size_t)IN_CH * HID_CH * 2);
    unsigned short* W2t  = (unsigned short*)p;  p += alignup((size_t)HID_CH * OUT_CH * 2);
    float*          t    = (float*)p;           p += alignup((size_t)N * HID_CH * 4);  // t1, then t2
    float*          h    = (float*)p;           p += alignup((size_t)N * HID_CH * 4);
    float*          z    = (float*)p;           p += alignup((size_t)N * OUT_CH * 4);
    (void)ws_size;

    const int B = 256;
    auto gr = [&](size_t total) { return (unsigned)((total + B - 1) / B); };

    // ---- normalization coefficients ----
    k_deg_init <<<gr(N), B, 0, stream>>>(dinv, N);
    k_deg_count<<<gr(E), B, 0, stream>>>(pei + E, dinv, E);
    k_dinv     <<<gr(N), B, 0, stream>>>(dinv, N);

    // ---- bf16 conversions ----
    k_cvt_bf16 <<<gr((size_t)N * IN_CH), B, 0, stream>>>(x, xb, (size_t)N * IN_CH);
    k_cvt_wT   <<<gr(IN_CH * HID_CH), B, 0, stream>>>(W1, W1t, IN_CH, HID_CH);
    k_cvt_wT   <<<gr(HID_CH * OUT_CH), B, 0, stream>>>(W2, W2t, HID_CH, OUT_CH);

    // ---- layer 1: t1 = x @ W1 (WMMA), aggregate, +b1, relu ----
    k_gemm_bf16<<<(N + 63) / 64, 32 * (HID_CH / 32), 0, stream>>>(xb, W1t, t, N, IN_CH, HID_CH);
    k_agg_init <<<gr((size_t)N * HID_CH), B, 0, stream>>>(t, dinv, b1, h, N, HID_CH, 7);
    k_scatter  <<<gr((size_t)E * HID_CH), B, 0, stream>>>(t, dinv, pei, h, E, HID_CH, 7);
    k_relu_cvt <<<gr((size_t)N * HID_CH), B, 0, stream>>>(h, xb, (size_t)N * HID_CH); // xb := hb

    // ---- layer 2: t2 = h @ W2 (WMMA), aggregate, +b2 ----
    k_gemm_bf16<<<(N + 63) / 64, 32 * (OUT_CH / 32), 0, stream>>>(xb, W2t, t, N, HID_CH, OUT_CH);
    k_agg_init <<<gr((size_t)N * OUT_CH), B, 0, stream>>>(t, dinv, b2, z, N, OUT_CH, 6);
    k_scatter  <<<gr((size_t)E * OUT_CH), B, 0, stream>>>(t, dinv, pei, z, E, OUT_CH, 6);

    // ---- decode ----
    float* out = (float*)d_out;
    k_decode   <<<gr(E), B, 0, stream>>>(z, pei, out, E);
    k_decode   <<<gr(E), B, 0, stream>>>(z, nei, out + E, E);
    (void)out_size; (void)n_in;
}
